// HierarchicalPlasticRNN_17806934409699
// MI455X (gfx1250) — matrix operand order repaired and checked by reading, not compiled, not run on hardware
//
#include <hip/hip_runtime.h>
#include <math.h>

// ---------------- problem constants (match reference) ----------------
#define HDIM    2560
#define ESIZE   2048
#define INSZ    128
#define TSTEPS  32
#define BATCH   64
#define ALPHA_F 0.2f            // dt_x / tau_x = 0.02/0.1
#define SPLITK  8
#define KCHUNK  (HDIM / SPLITK) // 320 -> 10 wmma iters per wave

// ---------------- CDNA5 WMMA types ----------------
typedef __attribute__((ext_vector_type(16))) __bf16 v16bf;
typedef __attribute__((ext_vector_type(8)))  float  v8f;

union Frag { v16bf v; uint4 q[2]; };

__device__ __forceinline__ unsigned short f2bf(float f) {
  union { float f; unsigned u; } x; x.f = f;
  unsigned r = x.u + 0x7FFFu + ((x.u >> 16) & 1u);   // round-to-nearest-even
  return (unsigned short)(r >> 16);
}

__device__ __forceinline__ float retanh_f(float x) { return tanhf(fmaxf(x, 0.0f)); }

__device__ __forceinline__ int area_of(int i) {
  return (i < ESIZE) ? (i >> 9) : ((i - ESIZE) >> 7); // E: /512, I: /128
}

// ---------------- one-time prep kernels ----------------
// Wr_bf16[i,j] = bf16( mask_rec(i,j) * |W_rec[i,j]| ), mask computed analytically
__global__ void k_prep_wrec(const float* __restrict__ Wrec, unsigned short* __restrict__ Wr) {
  int idx = blockIdx.x * blockDim.x + threadIdx.x;
  if (idx >= HDIM * HDIM) return;
  int i = idx / HDIM, j = idx - i * HDIM;
  float w = fabsf(Wrec[idx]);
  float v = 0.0f;
  if (i != j) {
    int ai = area_of(i), aj = area_of(j);
    if (j < ESIZE) {                 // excitatory column: tridiagonal area graph, +
      int d = ai - aj;
      if (d >= -1 && d <= 1) v = w;
    } else {                         // inhibitory column: same area only, -
      if (ai == aj) v = -w;
    }
  }
  Wr[idx] = f2bf(v);
}

// Wi_bf16[i,j] = bf16( |W_in[i,j]| ) if row i is in area 0 (E rows 0..511, I rows 2048..2175)
__global__ void k_prep_win(const float* __restrict__ Win, unsigned short* __restrict__ Wi) {
  int idx = blockIdx.x * blockDim.x + threadIdx.x;
  if (idx >= HDIM * INSZ) return;
  int i = idx / INSZ;
  bool rcv = (i < 512) || (i >= ESIZE && i < ESIZE + 128);
  Wi[idx] = f2bf(rcv ? fabsf(Win[idx]) : 0.0f);
}

// xa[t,b,k] = bf16( relu(x[t,b,k]) )
__global__ void k_prep_x(const float* __restrict__ x, unsigned short* __restrict__ xa) {
  int idx = blockIdx.x * blockDim.x + threadIdx.x;
  if (idx >= TSTEPS * BATCH * INSZ) return;
  xa[idx] = f2bf(fmaxf(x[idx], 0.0f));
}

// state = state0 ; act = bf16(retanh(state0))
__global__ void k_init(const float* __restrict__ state0, float* __restrict__ state,
                       unsigned short* __restrict__ act) {
  int idx = blockIdx.x * blockDim.x + threadIdx.x;
  if (idx >= BATCH * HDIM) return;
  float s = state0[idx];
  state[idx] = s;
  act[idx] = f2bf(retanh_f(s));
}

// ---------------- WMMA tile helper ----------------
// C[16x16] += A[m0..m0+15, kbeg..kend) * B[n0..n0+15, kbeg..kend)^T
// A, Bm row-major with K contiguous. One wave (32 lanes), EXEC all ones.
__device__ __forceinline__ v8f wmma_tile(const unsigned short* __restrict__ A, int lda,
                                         const unsigned short* __restrict__ Bm, int ldb,
                                         int m0, int n0, int kbeg, int kend) {
  int lane = threadIdx.x & 31;
  int r    = lane & 15;        // A row / B column within tile
  int half = lane >> 4;
  // A 16x32 bf16 layout: lanes 0-15 hold K {k0..k0+7, k0+16..k0+23}; lanes 16-31 +8
  const unsigned short* ar = A  + (size_t)(m0 + r) * lda + (half << 3);
  // B 32x16 bf16 layout: lanes 0-15 hold K {k0..k0+7, k0+16..k0+23}; lanes 16-31 +8
  const unsigned short* br = Bm + (size_t)(n0 + r) * ldb + (half << 3);
  v8f c = {0.f, 0.f, 0.f, 0.f, 0.f, 0.f, 0.f, 0.f};
  for (int k = kbeg; k < kend; k += 32) {
    Frag a, b;
    a.q[0] = *(const uint4*)(ar + k);
    a.q[1] = *(const uint4*)(ar + k + 16);
    b.q[0] = *(const uint4*)(br + k);
    b.q[1] = *(const uint4*)(br + k + 16);
    c = __builtin_amdgcn_wmma_f32_16x16x32_bf16(false, a.v, false, b.v,
                                                (short)0, c, false, false);
  }
  return c;
}

// ---------------- batched input drive: inb[t*B+b, h] = relu(x)@Wi^T + b_rec ----------------
__global__ void k_inb(const unsigned short* __restrict__ xa,   // [T*B, 128] bf16
                      const unsigned short* __restrict__ Wi,   // [H, 128]   bf16
                      const float* __restrict__ brec,          // [H]
                      float* __restrict__ inb) {               // [T*B, H]
  int m0 = blockIdx.x << 4;   // over T*B = 2048
  int n0 = blockIdx.y << 4;   // over H = 2560
  v8f c = wmma_tile(xa, INSZ, Wi, INSZ, m0, n0, 0, INSZ);
  int lane = threadIdx.x & 31;
  int n  = n0 + (lane & 15);
  int mb = m0 + ((lane >> 4) << 3);
  float bb = brec[n];
#pragma unroll
  for (int r = 0; r < 8; ++r)
    inb[(size_t)(mb + r) * HDIM + n] = c[r] + bb;
}

// ---------------- per-step split-K recurrent GEMM: part[s,b,h] = act@Wr^T (K chunk s) ----------------
__global__ void k_step_gemm(const unsigned short* __restrict__ act, // [B, H] bf16
                            const unsigned short* __restrict__ Wr,  // [H, H] bf16
                            float* __restrict__ part) {             // [SPLITK, B, H]
  int m0 = blockIdx.x << 4;   // over B = 64  -> 4
  int n0 = blockIdx.y << 4;   // over H      -> 160
  int s  = blockIdx.z;        // split-K     -> 8
  v8f c = wmma_tile(act, HDIM, Wr, HDIM, m0, n0, s * KCHUNK, (s + 1) * KCHUNK);
  int lane = threadIdx.x & 31;
  int n  = n0 + (lane & 15);
  int mb = m0 + ((lane >> 4) << 3);
  float* p = part + (size_t)s * BATCH * HDIM;
#pragma unroll
  for (int r = 0; r < 8; ++r)
    p[(size_t)(mb + r) * HDIM + n] = c[r];
}

// ---------------- per-step reduce + leaky update ----------------
__global__ void k_update(const float* __restrict__ part,    // [SPLITK, B, H]
                         const float* __restrict__ inb_t,   // [B, H] (slice at t)
                         float* __restrict__ state,         // [B, H]
                         unsigned short* __restrict__ act,  // [B, H] bf16 (next-step input)
                         float* __restrict__ out_t) {       // [B, H] (outputs[t])
  int idx = blockIdx.x * blockDim.x + threadIdx.x;
  if (idx >= BATCH * HDIM) return;
  float tot = inb_t[idx];
#pragma unroll
  for (int s = 0; s < SPLITK; ++s) tot += part[(size_t)s * BATCH * HDIM + idx];
  float ns = state[idx] * (1.0f - ALPHA_F) + ALPHA_F * tot;
  state[idx] = ns;
  float o = retanh_f(ns);
  out_t[idx] = o;
  act[idx] = f2bf(o);
}

// ---------------- host orchestration ----------------
extern "C" void kernel_launch(void* const* d_in, const int* in_sizes, int n_in,
                              void* d_out, int out_size, void* d_ws, size_t ws_size,
                              hipStream_t stream) {
  (void)in_sizes; (void)n_in; (void)out_size; (void)ws_size;
  const float* x      = (const float*)d_in[0];  // [T, B, 128]
  const float* W_in   = (const float*)d_in[1];  // [H, 128]
  const float* W_rec  = (const float*)d_in[2];  // [H, H]
  const float* b_rec  = (const float*)d_in[3];  // [H]
  const float* state0 = (const float*)d_in[4];  // [B, H]
  float* out = (float*)d_out;                   // [T, B, H]

  // workspace carve-up (all chunks multiple of 256B)
  char* ws = (char*)d_ws;
  unsigned short* Wr   = (unsigned short*)ws;            ws += (size_t)HDIM * HDIM * 2;        // 13.1 MB
  unsigned short* Wi   = (unsigned short*)ws;            ws += (size_t)HDIM * INSZ * 2;        // 0.66 MB
  unsigned short* xa   = (unsigned short*)ws;            ws += (size_t)TSTEPS * BATCH * INSZ * 2; // 0.52 MB
  float*          inb  = (float*)ws;                     ws += (size_t)TSTEPS * BATCH * HDIM * 4; // 21.0 MB
  float*          st   = (float*)ws;                     ws += (size_t)BATCH * HDIM * 4;       // 0.66 MB
  unsigned short* act  = (unsigned short*)ws;            ws += (size_t)BATCH * HDIM * 2;       // 0.33 MB
  float*          part = (float*)ws;                     /* SPLITK*B*H*4 = 5.2 MB */

  // one-time prep
  k_prep_wrec<<<(HDIM * HDIM + 255) / 256, 256, 0, stream>>>(W_rec, Wr);
  k_prep_win <<<(HDIM * INSZ + 255) / 256, 256, 0, stream>>>(W_in, Wi);
  k_prep_x   <<<(TSTEPS * BATCH * INSZ + 255) / 256, 256, 0, stream>>>(x, xa);
  k_init     <<<(BATCH * HDIM + 255) / 256, 256, 0, stream>>>(state0, st, act);

  // batched input drive for all timesteps (WMMA, K=128)
  k_inb<<<dim3((TSTEPS * BATCH) / 16, HDIM / 16), 32, 0, stream>>>(xa, Wi, b_rec, inb);

  // sequential recurrence: split-K WMMA GEMM + reduce/update per step
  for (int t = 0; t < TSTEPS; ++t) {
    k_step_gemm<<<dim3(BATCH / 16, HDIM / 16, SPLITK), 32, 0, stream>>>(act, Wr, part);
    k_update<<<(BATCH * HDIM + 255) / 256, 256, 0, stream>>>(
        part, inb + (size_t)t * BATCH * HDIM, st, act, out + (size_t)t * BATCH * HDIM);
  }
}
